// RodriNet_66340064854459
// MI455X (gfx1250) — matrix operand order repaired and checked by reading, not compiled, not run on hardware
//
#include <hip/hip_runtime.h>
#include <hip/hip_bf16.h>

// RodriNet fused forward for MI455X (gfx1250, wave32, WMMA).
// One workgroup = 8 batch rows = 256 token-rows. 8 waves; wave w owns token
// rows [32w, 32w+32) == batch row w of the tile, making the whole attention
// pipeline wave-local. All dense GEMMs (encoder included) use
// v_wmma_f32_16x16x32_f16.

typedef __attribute__((ext_vector_type(16))) _Float16 v16h;
typedef __attribute__((ext_vector_type(8)))  float    v8f;

namespace {
constexpr int kL   = 32;   // links
constexpr int kJ   = 31;   // joints
constexpr int kCJ  = 8;
constexpr int kNH  = 4;
constexpr int kNB  = 3;
constexpr int kOBS = 128;
constexpr int kTOK = 64;
constexpr int kBT  = 8;                      // batch rows per block
constexpr int kROWS = kBT * kL;              // 256
constexpr int kOUT  = kL * kTOK + kJ * kCJ;  // 2296
}

__device__ __forceinline__ v8f wmma_f16(v16h a, v16h b, v8f c) {
  // v_wmma_f32_16x16x32_f16  (neg_a, A, neg_b, B, c_mod, C, reuse_a, reuse_b)
  return __builtin_amdgcn_wmma_f32_16x16x32_f16(false, a, false, b, (short)0, c,
                                                false, false);
}

// ---- fragment builders, wave32; A and B both use k = 16*(lane/16) + e ----

// A (16x32 f16) from row-major LDS matrix, leading dim lda, tile at (row0,k0).
// lanes 0-15: M=lane, halves e=0..15 -> K=e; lanes 16-31: K=16+e.
__device__ __forceinline__ v16h frag_a(const _Float16* A, int lda, int row0,
                                       int k0, int lane) {
  v16h a;
  const _Float16* p = A + (row0 + (lane & 15)) * lda + k0 + ((lane & 16) ? 16 : 0);
#pragma unroll
  for (int e = 0; e < 16; ++e) a[e] = p[e];
  return a;
}

// A for the encoder: obs tile with only 8 real rows (row = lane&7 duplicates).
__device__ __forceinline__ v16h frag_a_obs(const _Float16* O, int k0, int lane) {
  v16h a;
  const _Float16* p = O + (lane & 7) * kOBS + k0 + ((lane & 16) ? 16 : 0);
#pragma unroll
  for (int e = 0; e < 16; ++e) a[e] = p[e];
  return a;
}

// A (16x16 padded to K=32 with zeros) from row-major [rows][16] LDS (q tiles).
__device__ __forceinline__ v16h frag_a_pad(const _Float16* A, int row0, int lane) {
  v16h a;
  const _Float16* p = A + (row0 + (lane & 15)) * 16;
  int ko = (lane & 16) ? 16 : 0;
#pragma unroll
  for (int e = 0; e < 16; ++e) {
    int k = ko + e;
    a[e] = (k < 16) ? p[k] : (_Float16)0.0f;
  }
  return a;
}

// B (32x16, B[k][n] = W[n][k]) from a row-major f32 weight (out_dim x in_dim).
// lane%16 = N column; lanes 0-15 hold K=0..15 (e), lanes 16-31 K=16..31.
__device__ __forceinline__ v16h frag_b_wt(const float* W, int ldw, int n0,
                                          int k0, int lane) {
  v16h b;
  const float* p = W + (n0 + (lane & 15)) * ldw + k0 + ((lane & 16) ? 16 : 0);
#pragma unroll
  for (int e = 0; e < 16; ++e) b[e] = (_Float16)p[e];
  return b;
}

// B (32x16) from row-major f16 [K][16] LDS (v tiles: B[k][n] = V[k][n]).
__device__ __forceinline__ v16h frag_b_kn16(const _Float16* Bm, int lane) {
  v16h b;
  const _Float16* p = Bm + (((lane & 16) ? 16 : 0)) * 16 + (lane & 15);
#pragma unroll
  for (int e = 0; e < 16; ++e) b[e] = p[e * 16];
  return b;
}

// B for scores: B[k][n] = K[n0+n][k], K rows are tokens, HD=16 zero-padded.
__device__ __forceinline__ v16h frag_b_ktpad(const _Float16* Km, int n0, int lane) {
  v16h b;
  const _Float16* p = Km + (n0 + (lane & 15)) * 16;
  int ko = (lane & 16) ? 16 : 0;
#pragma unroll
  for (int e = 0; e < 16; ++e) {
    int k = ko + e;
    b[e] = (k < 16) ? p[k] : (_Float16)0.0f;
  }
  return b;
}

// GEMM: out[rows][*] = A[rows][64] @ W^T + bias, f16 out.
// A-frags hoisted; N-tiles outermost so each global B-fragment is loaded once
// and feeds two independent accumulator chains (c0/c1) for XDL pipelining.
__device__ __forceinline__ void gemm_wave(const _Float16* A, const float* W,
                                          const float* bias, _Float16* out,
                                          int ldo, int ntiles, int mrow0,
                                          int lane) {
  v16h a00 = frag_a(A, 64, mrow0, 0, lane);
  v16h a01 = frag_a(A, 64, mrow0, 32, lane);
  v16h a10 = frag_a(A, 64, mrow0 + 16, 0, lane);
  v16h a11 = frag_a(A, 64, mrow0 + 16, 32, lane);
  const int mo = (lane & 16) ? 8 : 0, n = lane & 15;
  for (int nt = 0; nt < ntiles; ++nt) {
    int n0 = nt * 16;
    v16h b0 = frag_b_wt(W, 64, n0, 0, lane);
    v16h b1 = frag_b_wt(W, 64, n0, 32, lane);
    float bv = bias[n0 + n];
    v8f c0, c1;
#pragma unroll
    for (int r = 0; r < 8; ++r) { c0[r] = bv; c1[r] = bv; }
    c0 = wmma_f16(a00, b0, c0);
    c1 = wmma_f16(a10, b0, c1);
    c0 = wmma_f16(a01, b1, c0);
    c1 = wmma_f16(a11, b1, c1);
#pragma unroll
    for (int r = 0; r < 8; ++r) {
      out[(mrow0 + mo + r) * ldo + n0 + n]      = (_Float16)c0[r];
      out[(mrow0 + 16 + mo + r) * ldo + n0 + n] = (_Float16)c1[r];
    }
  }
}

__global__ __launch_bounds__(256, 1) void rodrinet_fused(
    const float* __restrict__ obs, const float* __restrict__ link_enc_w1,
    const float* __restrict__ link_enc_b1, const float* __restrict__ link_enc_w2,
    const float* __restrict__ link_enc_b2, const float* __restrict__ joint_enc_w,
    const float* __restrict__ joint_enc_b, const float* __restrict__ rod_mix,
    const float* __restrict__ rod_theta_w, const float* __restrict__ rod_theta_b,
    const float* __restrict__ rod_gen, const float* __restrict__ rod_ln_g,
    const float* __restrict__ rod_ln_b, const float* __restrict__ jl_w,
    const float* __restrict__ jl_b, const float* __restrict__ proj_in_w,
    const float* __restrict__ proj_in_b, const float* __restrict__ qkv_w,
    const float* __restrict__ qkv_b, const float* __restrict__ attn_out_w,
    const float* __restrict__ attn_out_b, const float* __restrict__ proj_out_w,
    const float* __restrict__ proj_out_b, const float* __restrict__ attn_ln_g,
    const float* __restrict__ attn_ln_b, const int* __restrict__ p_idx,
    const int* __restrict__ c_idx, float* __restrict__ out) {
  __shared__ _Float16 s_tok[kROWS * kTOK];   // link tokens (f16)
  __shared__ _Float16 s_x[kROWS * kTOK];     // proj_in / attn_out result
  __shared__ _Float16 s_o[kROWS * kTOK];     // attention output (per-head cols)
  __shared__ _Float16 s_q[kROWS * 16];
  __shared__ _Float16 s_k[kROWS * 16];
  __shared__ _Float16 s_v[kROWS * 16];
  __shared__ _Float16 s_p[8 * 32 * 32];      // softmax probs / encoder h staging
  __shared__ _Float16 s_obs16[kBT * kOBS];   // f16 obs for encoder WMMA
  __shared__ float    s_joint[kBT * kJ * kCJ];
  __shared__ float    s_obs[kBT * kOBS];

  const int tid = threadIdx.x;
  const int lane = tid & 31;
  const int wave = tid >> 5;
  const int mrow0 = wave * 32;               // wave-local token rows == batch b=wave
  const int b0 = blockIdx.x * kBT;           // global batch base

  // ---------------- stage obs (f32 for VALU, f16 for WMMA) ----------------
  for (int i = tid; i < kBT * kOBS; i += 256) {
    float v = obs[(b0 + i / kOBS) * kOBS + (i % kOBS)];
    s_obs[i] = v;
    s_obs16[i] = (_Float16)v;
  }
  __syncthreads();

  // ---------------- link encoder via WMMA: wave owns 4 links ----------------
  {
    _Float16* s_h = s_p + wave * (16 * kTOK);  // per-wave 16x64 h staging
    const int mo = (lane & 16) ? 8 : 0, n = lane & 15;
    for (int il = 0; il < 4; ++il) {
      int l = wave * 4 + il;
      // h = elu(obs @ w1_l^T + b1): M=16 (8 real), N=64, K=128
      const float* w1 = link_enc_w1 + l * 64 * kOBS;
      const float* b1 = link_enc_b1 + l * 64;
#pragma unroll
      for (int nt = 0; nt < 4; ++nt) {
        float bv = b1[nt * 16 + n];
        v8f c;
#pragma unroll
        for (int r = 0; r < 8; ++r) c[r] = bv;
#pragma unroll
        for (int kc = 0; kc < 4; ++kc)
          c = wmma_f16(frag_a_obs(s_obs16, kc * 32, lane),
                       frag_b_wt(w1, kOBS, nt * 16, kc * 32, lane), c);
#pragma unroll
        for (int r = 0; r < 8; ++r) {
          float v = c[r];
          v = (v > 0.f) ? v : (__expf(v) - 1.f);  // ELU
          s_h[(mo + r) * kTOK + nt * 16 + n] = (_Float16)v;
        }
      }
      // link = h @ w2_l^T + b2: M=16 (8 real), N=64, K=64
      const float* w2 = link_enc_w2 + l * kTOK * 64;
      const float* b2 = link_enc_b2 + l * kTOK;
      v16h a0 = frag_a(s_h, 64, 0, 0, lane);
      v16h a1 = frag_a(s_h, 64, 0, 32, lane);
#pragma unroll
      for (int nt = 0; nt < 4; ++nt) {
        float bv = b2[nt * 16 + n];
        v8f c;
#pragma unroll
        for (int r = 0; r < 8; ++r) c[r] = bv;
        c = wmma_f16(a0, frag_b_wt(w2, 64, nt * 16, 0, lane), c);
        c = wmma_f16(a1, frag_b_wt(w2, 64, nt * 16, 32, lane), c);
        if (!(lane & 16)) {  // rows 8..15 are padding duplicates
#pragma unroll
          for (int r = 0; r < 8; ++r)
            s_tok[(r * kL + l) * kTOK + nt * 16 + n] = (_Float16)c[r];
        }
      }
    }
  }
  if (tid < kBT * kJ) {  // joint encoder (small: ~1K VALU cycles, co-executes)
    int b = tid / kJ, j = tid % kJ;
    for (int q = 0; q < kCJ; ++q) {
      float acc = joint_enc_b[j * kCJ + q];
      const float* w = joint_enc_w + (j * kCJ + q) * kOBS;
      for (int o = 0; o < kOBS; ++o) acc += s_obs[b * kOBS + o] * w[o];
      s_joint[(b * kJ + j) * kCJ + q] = acc;
    }
  }
  __syncthreads();

  // ---------------- NB iterations ----------------
  for (int t = 0; t < kNB; ++t) {
    // prefetch this iteration's weights into L2 (global_prefetch_b8)
    {
      const float* pf = qkv_w + t * 192 * 64;
      for (int i = tid * 16; i < 192 * 64; i += 256 * 16) __builtin_prefetch(pf + i);
      if (tid * 16 < 64 * 64) {
        __builtin_prefetch(proj_in_w + t * 4096 + tid * 16);
        __builtin_prefetch(attn_out_w + t * 4096 + tid * 16);
        __builtin_prefetch(proj_out_w + t * 4096 + tid * 16);
      }
    }

    // ---- rod / Rodrigues / LN / joint phase (VALU, functional semantics) ----
    {
      float comb[64];
      int bb, ci;
      if (tid < kBT * kJ) {
        int jj = tid % kJ;
        bb = tid / kJ;
        int p = p_idx[jj];
        ci = c_idx[jj];
        const _Float16* tokb = s_tok + bb * (kL * kTOK);
        float Fp[64];
        for (int e = 0; e < 64; ++e) Fp[e] = (float)tokb[p * 64 + e];
        for (int e = 0; e < 64; ++e) comb[e] = (float)tokb[ci * 64 + e];  // Fc
        const float* mixw = rod_mix + ((t * kJ + jj) * 4) * 4;
        const float* thw  = rod_theta_w + ((t * kJ + jj) * 4) * kCJ;
        const float* thb  = rod_theta_b + (t * kJ + jj) * 4;
        const float* G    = rod_gen + (t * kJ + jj) * 16;
        float jt[8];
        for (int q = 0; q < kCJ; ++q) jt[q] = s_joint[(bb * kJ + jj) * kCJ + q];
        for (int o = 0; o < 4; ++o) {
          float Fm[16];
          for (int e = 0; e < 16; ++e) {
            float acc = 0.f;
            for (int cc = 0; cc < 4; ++cc) acc += mixw[o * 4 + cc] * Fp[cc * 16 + e];
            Fm[e] = acc;
          }
          float ang = thb[o];
          for (int q = 0; q < kCJ; ++q) ang += jt[q] * thw[o * kCJ + q];
          float GF[16], GGF[16];
          for (int x = 0; x < 4; ++x)
            for (int y = 0; y < 4; ++y) {
              float acc = 0.f;
              for (int z = 0; z < 4; ++z) acc += G[x * 4 + z] * Fm[z * 4 + y];
              GF[x * 4 + y] = acc;
            }
          for (int x = 0; x < 4; ++x)
            for (int y = 0; y < 4; ++y) {
              float acc = 0.f;
              for (int z = 0; z < 4; ++z) acc += G[x * 4 + z] * GF[z * 4 + y];
              GGF[x * 4 + y] = acc;
            }
          float sn = __sinf(ang), c1 = 1.f - __cosf(ang);
          for (int e = 0; e < 16; ++e)
            comb[o * 16 + e] += Fm[e] + sn * GF[e] + c1 * GGF[e];
        }
      } else {  // 8 root threads: LN of link[:,0]
        bb = tid - kBT * kJ;
        ci = 0;
        const _Float16* tokb = s_tok + bb * (kL * kTOK);
        for (int e = 0; e < 64; ++e) comb[e] = (float)tokb[e];
      }
      // LayerNorm over 64 with per-(link,elem) gain/bias
      {
        float mu = 0.f;
        for (int e = 0; e < 64; ++e) mu += comb[e];
        mu *= (1.f / 64.f);
        float var = 0.f;
        for (int e = 0; e < 64; ++e) { float d = comb[e] - mu; var += d * d; }
        var *= (1.f / 64.f);
        float inv = rsqrtf(var + 1e-5f);
        const float* g  = rod_ln_g + (t * kL + ci) * kTOK;
        const float* bt = rod_ln_b + (t * kL + ci) * kTOK;
        for (int e = 0; e < 64; ++e) comb[e] = (comb[e] - mu) * inv * g[e] + bt[e];
      }
      __syncthreads();  // all reads of old link done
      {
        _Float16* tokb = s_tok + bb * (kL * kTOK);
        for (int e = 0; e < 64; ++e) tokb[ci * 64 + e] = (_Float16)comb[e];
      }
      __syncthreads();  // new link visible
      if (tid < kBT * kJ) {  // joint += child_flat @ jl_w^T + jl_b
        int jj = tid % kJ;
        const _Float16* ch = s_tok + bb * (kL * kTOK) + ci * 64;
        const float* w  = jl_w + ((t * kJ + jj) * kCJ) * kTOK;
        const float* jb = jl_b + (t * kJ + jj) * kCJ;
        for (int q = 0; q < kCJ; ++q) {
          float acc = jb[q] + s_joint[(bb * kJ + jj) * kCJ + q];
          for (int e = 0; e < 64; ++e) acc += (float)ch[e] * w[q * 64 + e];
          s_joint[(bb * kJ + jj) * kCJ + q] = acc;
        }
      }
      __syncthreads();
    }

    // ---- attention block: entirely wave-local from here ----
    // x = tok @ proj_in_w^T + b
    gemm_wave(s_tok, proj_in_w + t * 64 * 64, proj_in_b + t * 64, s_x, 64, 4,
              mrow0, lane);

    const float* Wqkv = qkv_w + t * 192 * 64;
    const float* Bqkv = qkv_b + t * 192;
    for (int h = 0; h < kNH; ++h) {
      // per-head q,k,v slices (N=16 each)
      gemm_wave(s_x, Wqkv + (h * 16) * 64,        Bqkv + h * 16,        s_q, 16, 1, mrow0, lane);
      gemm_wave(s_x, Wqkv + (64 + h * 16) * 64,   Bqkv + 64 + h * 16,   s_k, 16, 1, mrow0, lane);
      gemm_wave(s_x, Wqkv + (128 + h * 16) * 64,  Bqkv + 128 + h * 16,  s_v, 16, 1, mrow0, lane);

      const _Float16* qb = s_q + mrow0 * 16;
      const _Float16* kb = s_k + mrow0 * 16;
      const _Float16* vb = s_v + mrow0 * 16;
      _Float16* pb = s_p + wave * (32 * 32);

      // scores: 32x32 = (2x2) WMMA tiles, K=16 zero-padded to 32
      v8f cs[2][2];
#pragma unroll
      for (int mt = 0; mt < 2; ++mt) {
        v16h a = frag_a_pad(qb, mt * 16, lane);
#pragma unroll
        for (int nt = 0; nt < 2; ++nt) {
          v16h bk = frag_b_ktpad(kb, nt * 16, lane);
          v8f c;
#pragma unroll
          for (int r = 0; r < 8; ++r) c[r] = 0.f;
          cs[mt][nt] = wmma_f16(a, bk, c);
        }
      }
      // softmax over rows of 32 (row = 16-lane half; reduce with shfl_xor)
      {
        int mo = (lane & 16) ? 8 : 0, n = lane & 15;
#pragma unroll
        for (int mt = 0; mt < 2; ++mt) {
#pragma unroll
          for (int r = 0; r < 8; ++r) {
            float x0 = cs[mt][0][r] * 0.25f;  // 1/sqrt(HD=16)
            float x1 = cs[mt][1][r] * 0.25f;
            float m = fmaxf(x0, x1);
            for (int d = 1; d < 16; d <<= 1) m = fmaxf(m, __shfl_xor(m, d));
            float e0 = __expf(x0 - m), e1 = __expf(x1 - m);
            float s = e0 + e1;
            for (int d = 1; d < 16; d <<= 1) s += __shfl_xor(s, d);
            float inv = 1.f / s;
            int row = mt * 16 + mo + r;
            pb[row * 32 + n]      = (_Float16)(e0 * inv);
            pb[row * 32 + 16 + n] = (_Float16)(e1 * inv);
          }
        }
      }
      // o_head = probs @ v  (M=32, K=32, N=16) -> columns [h*16, h*16+16)
#pragma unroll
      for (int mt = 0; mt < 2; ++mt) {
        v16h a = frag_a(pb, 32, mt * 16, 0, lane);
        v16h bv = frag_b_kn16(vb, lane);
        v8f c;
#pragma unroll
        for (int r = 0; r < 8; ++r) c[r] = 0.f;
        c = wmma_f16(a, bv, c);
        int mo = (lane & 16) ? 8 : 0, n = lane & 15;
#pragma unroll
        for (int r = 0; r < 8; ++r)
          s_o[(mrow0 + mt * 16 + mo + r) * 64 + h * 16 + n] = (_Float16)c[r];
      }
    }

    // o @ attn_out_w^T + b  -> s_x (reuse)
    gemm_wave(s_o, attn_out_w + t * 64 * 64, attn_out_b + t * 64, s_x, 64, 4,
              mrow0, lane);

    // proj_out + residual + LayerNorm, fused in registers
    {
      const float* Wp = proj_out_w + t * 64 * 64;
      const float* bp = proj_out_b + t * 64;
      const float* lg = attn_ln_g + t * kTOK;
      const float* lb = attn_ln_b + t * kTOK;
      int mo = (lane & 16) ? 8 : 0, n = lane & 15;
#pragma unroll
      for (int mt = 0; mt < 2; ++mt) {
        int r0 = mrow0 + mt * 16;
        v16h a0 = frag_a(s_x, 64, r0, 0, lane);
        v16h a1 = frag_a(s_x, 64, r0, 32, lane);
        v8f c[4];
#pragma unroll
        for (int nt = 0; nt < 4; ++nt) {
          float bv = bp[nt * 16 + n];
#pragma unroll
          for (int r = 0; r < 8; ++r) c[nt][r] = bv;
          c[nt] = wmma_f16(a0, frag_b_wt(Wp, 64, nt * 16, 0, lane), c[nt]);
          c[nt] = wmma_f16(a1, frag_b_wt(Wp, 64, nt * 16, 32, lane), c[nt]);
        }
#pragma unroll
        for (int r = 0; r < 8; ++r) {
          int row = r0 + mo + r;
#pragma unroll
          for (int nt = 0; nt < 4; ++nt)
            c[nt][r] = c[nt][r] + (float)s_tok[row * 64 + nt * 16 + n];  // residual
          float s1 = c[0][r] + c[1][r] + c[2][r] + c[3][r];
          for (int d = 1; d < 16; d <<= 1) s1 += __shfl_xor(s1, d);
          float mu = s1 * (1.f / 64.f);
          float sq = 0.f;
#pragma unroll
          for (int nt = 0; nt < 4; ++nt) {
            float dd = c[nt][r] - mu;
            sq += dd * dd;
          }
          for (int d = 1; d < 16; d <<= 1) sq += __shfl_xor(sq, d);
          float inv = rsqrtf(sq * (1.f / 64.f) + 1e-5f);
#pragma unroll
          for (int nt = 0; nt < 4; ++nt) {
            int col = nt * 16 + n;
            s_tok[row * 64 + col] =
                (_Float16)((c[nt][r] - mu) * inv * lg[col] + lb[col]);
          }
        }
      }
    }
    __syncthreads();  // next rod phase / epilogue reads cross-wave rows
  }

  // ---------------- epilogue: concat(link, joint) ----------------
  for (int i = tid; i < kBT * kOUT; i += 256) {
    int b = i / kOUT, r = i % kOUT;
    float v = (r < kL * kTOK) ? (float)s_tok[b * (kL * kTOK) + r]
                              : s_joint[b * (kJ * kCJ) + (r - kL * kTOK)];
    out[(size_t)(b0 + b) * kOUT + r] = v;
  }
}

extern "C" void kernel_launch(void* const* d_in, const int* in_sizes, int n_in,
                              void* d_out, int out_size, void* d_ws, size_t ws_size,
                              hipStream_t stream) {
  (void)n_in; (void)d_ws; (void)ws_size; (void)out_size;
  const int B = in_sizes[0] / kOBS;
  dim3 grid(B / kBT), block(256);
  rodrinet_fused<<<grid, block, 0, stream>>>(
      (const float*)d_in[0],  (const float*)d_in[1],  (const float*)d_in[2],
      (const float*)d_in[3],  (const float*)d_in[4],  (const float*)d_in[5],
      (const float*)d_in[6],  (const float*)d_in[7],  (const float*)d_in[8],
      (const float*)d_in[9],  (const float*)d_in[10], (const float*)d_in[11],
      (const float*)d_in[12], (const float*)d_in[13], (const float*)d_in[14],
      (const float*)d_in[15], (const float*)d_in[16], (const float*)d_in[17],
      (const float*)d_in[18], (const float*)d_in[19], (const float*)d_in[20],
      (const float*)d_in[21], (const float*)d_in[22], (const float*)d_in[23],
      (const float*)d_in[24], (const int*)d_in[25],   (const int*)d_in[26],
      (float*)d_out);
}
